// SetAbstraction_89163521065356
// MI455X (gfx1250) — matrix operand order repaired and checked by reading, not compiled, not run on hardware
//
#include <hip/hip_runtime.h>

// PointNet++ SetAbstraction for MI455X (gfx1250, wave32).
// Pipeline: zero_stats -> fps -> knn+gather(bf16 feat, K padded 67->96)
//   -> WMMA gemm1 (+stats) -> finalize BN1 -> WMMA gemm2 (BN1+ReLU fused in staging, +stats)
//   -> finalize BN2 -> WMMA gemm3 (+stats) -> finalize BN3 -> maxpool (BN3 after max; gamma>0)

typedef __attribute__((ext_vector_type(16))) __bf16 v16bf;
typedef __attribute__((ext_vector_type(8)))  float  v8f;

#define NB    8
#define NPT   8192
#define NP    1024
#define NS    32
#define MROWS (NB * NP * NS)   // 262144
#define K1PAD 96               // 67 padded to 3 K-tiles of 32
#define CIN   64
#define C1    64
#define C2    64
#define C3    128

// ---------------------------------------------------------------- FPS
__global__ __launch_bounds__(1024) void fps_kernel(const float* __restrict__ xyz,
                                                   float* __restrict__ out_xyz,
                                                   float* __restrict__ new_xyz_t) {
  const int b = blockIdx.x;
  const int tid = threadIdx.x;
  const float* xb = xyz + (size_t)b * 3 * NPT;
  float px[8], py[8], pz[8], dist[8];
#pragma unroll
  for (int j = 0; j < 8; ++j) {
    int i = tid + 1024 * j;
    px[j] = xb[i]; py[j] = xb[NPT + i]; pz[j] = xb[2 * NPT + i];
    dist[j] = 1e10f;
  }
  __shared__ float s_bd[32];
  __shared__ int   s_bi[32];
  __shared__ float s_c[3];
  __shared__ int   s_far;
  int far = 0;
  for (int p = 0; p < NP; ++p) {
    int owner = far & 1023, oj = far >> 10;
    if (tid == owner) {
      float ox = px[0], oy = py[0], oz = pz[0];
#pragma unroll
      for (int j = 1; j < 8; ++j) if (oj == j) { ox = px[j]; oy = py[j]; oz = pz[j]; }
      s_c[0] = ox; s_c[1] = oy; s_c[2] = oz;
      out_xyz[(size_t)b * 3 * NP + 0 * NP + p] = ox;
      out_xyz[(size_t)b * 3 * NP + 1 * NP + p] = oy;
      out_xyz[(size_t)b * 3 * NP + 2 * NP + p] = oz;
      float* nt = new_xyz_t + ((size_t)b * NP + p) * 3;
      nt[0] = ox; nt[1] = oy; nt[2] = oz;
    }
    __syncthreads();
    const float cx = s_c[0], cy = s_c[1], cz = s_c[2];
    float bd = -1.0f; int bi = 0;
#pragma unroll
    for (int j = 0; j < 8; ++j) {
      float dx = px[j] - cx, dy = py[j] - cy, dz = pz[j] - cz;
      float d = dx * dx + dy * dy + dz * dz;
      dist[j] = fminf(dist[j], d);
      if (dist[j] > bd) { bd = dist[j]; bi = tid + 1024 * j; }
    }
#pragma unroll
    for (int m = 16; m > 0; m >>= 1) {                 // wave32 argmax
      float od = __shfl_xor(bd, m, 32);
      int   oi = __shfl_xor(bi, m, 32);
      if (od > bd || (od == bd && oi < bi)) { bd = od; bi = oi; }
    }
    if ((tid & 31) == 0) { s_bd[tid >> 5] = bd; s_bi[tid >> 5] = bi; }
    __syncthreads();
    if (tid < 32) {
      bd = s_bd[tid]; bi = s_bi[tid];
#pragma unroll
      for (int m = 16; m > 0; m >>= 1) {
        float od = __shfl_xor(bd, m, 32);
        int   oi = __shfl_xor(bi, m, 32);
        if (od > bd || (od == bd && oi < bi)) { bd = od; bi = oi; }
      }
      if (tid == 0) s_far = bi;
    }
    __syncthreads();
    far = s_far;
  }
}

// ------------------------------------------------------ KNN + gather (bf16 feat)
__global__ __launch_bounds__(256) void knn_gather_kernel(const float* __restrict__ xyz,
                                                         const float* __restrict__ points,
                                                         const float* __restrict__ new_xyz_t,
                                                         __bf16* __restrict__ feat) {
  const int b   = blockIdx.x >> 2;
  const int p   = ((blockIdx.x & 3) << 8) + threadIdx.x;
  const int tid = threadIdx.x;
  const float* xb = xyz + (size_t)b * 3 * NPT;
  const float* q  = new_xyz_t + ((size_t)b * NP + p) * 3;
  const float qx = q[0], qy = q[1], qz = q[2];

  float bd[NS]; int bi[NS];
#pragma unroll
  for (int j = 0; j < NS; ++j) { bd[j] = 3.4e38f; bi[j] = 0; }
  float curmax = 3.4e38f; int slot = 0;

  __shared__ float sx[2048], sy[2048], sz[2048];
  for (int t = 0; t < 4; ++t) {
    __syncthreads();
    for (int i = tid; i < 2048; i += 256) {
      sx[i] = xb[t * 2048 + i];
      sy[i] = xb[NPT + t * 2048 + i];
      sz[i] = xb[2 * NPT + t * 2048 + i];
    }
    __syncthreads();
    for (int i = 0; i < 2048; ++i) {
      float dx = sx[i] - qx, dy = sy[i] - qy, dz = sz[i] - qz;
      float d = dx * dx + dy * dy + dz * dz;
      if (d < curmax) {                                 // rare replacement path
        int idx = t * 2048 + i;
#pragma unroll
        for (int j = 0; j < NS; ++j) if (j == slot) { bd[j] = d; bi[j] = idx; }
        curmax = -1.0f;
#pragma unroll
        for (int j = 0; j < NS; ++j) if (bd[j] > curmax) { curmax = bd[j]; slot = j; }
      }
    }
  }
  const float* pb = points + (size_t)b * CIN * NPT;
  __bf16* rowbase = feat + ((size_t)b * NP + p) * NS * K1PAD;
#pragma unroll
  for (int s = 0; s < NS; ++s) {
    int idx = bi[s];
    __bf16* row = rowbase + s * K1PAD;
    row[0] = (__bf16)(xb[idx] - qx);
    row[1] = (__bf16)(xb[NPT + idx] - qy);
    row[2] = (__bf16)(xb[2 * NPT + idx] - qz);
    for (int c = 0; c < CIN; ++c) row[3 + c] = (__bf16)pb[(size_t)c * NPT + idx];
    for (int c = 3 + CIN; c < K1PAD; ++c) row[c] = (__bf16)0.0f;
  }
}

// ---------------------------------------------------------------- WMMA helpers
// A 16x32 bf16 frag: lane<16 -> K{0..15}, lane>=16 -> K{16..31}
__device__ __forceinline__ v16bf load_a_frag(const __bf16* X, int ldk, int row, int kbase) {
  union { uint4 q[2]; v16bf v; } f;
  const uint4* p0 = (const uint4*)(X + row * ldk + kbase);
  f.q[0] = p0[0];
  f.q[1] = p0[1];   // +8 elements (=16B)
  return f.v;
}
// B 32x16 bf16 frag: col = lane&15, K half selected by lane>>4 (kbase includes hi*16)
__device__ __forceinline__ v16bf load_b_frag(const __bf16* W, int ldk, int col, int kbase) {
  return *(const v16bf*)(W + col * ldk + kbase);
}

// ---------------------------------------------------------------- GEMM1 (K=96 padded)
__global__ __launch_bounds__(256) void gemm1_kernel(const __bf16* __restrict__ feat,
                                                    const float* __restrict__ w1,
                                                    float* __restrict__ h1,
                                                    float* __restrict__ stats) {
  __shared__ __attribute__((aligned(32))) __bf16 sX[128 * K1PAD];
  __shared__ __attribute__((aligned(32))) __bf16 sW[C1 * K1PAD];
  __shared__ float sSum[C1], sSq[C1];
  const int tid = threadIdx.x;
  const size_t m0 = (size_t)blockIdx.x * 128;

  const uint4* src = (const uint4*)(feat + m0 * K1PAD);
  uint4* dst = (uint4*)sX;
  for (int i = tid; i < 128 * K1PAD / 8; i += 256) dst[i] = src[i];
  for (int i = tid; i < C1 * K1PAD; i += 256) {
    int r = i / K1PAD, c = i % K1PAD;
    sW[i] = (c < 67) ? (__bf16)w1[r * 67 + c] : (__bf16)0.0f;   // bias dropped (BN cancels)
  }
  if (tid < C1) { sSum[tid] = 0.0f; sSq[tid] = 0.0f; }
  if (tid == 0 && blockIdx.x + 1 < gridDim.x)
    __builtin_prefetch(feat + (m0 + 128) * K1PAD, 0, 0);
  __syncthreads();

  const int lane = tid & 31, wave = tid >> 5;
  const int lo = lane & 15, hi = lane >> 4;
  const int arow = wave * 16 + lo;
  v8f acc[4];
#pragma unroll
  for (int nt = 0; nt < 4; ++nt) acc[nt] = (v8f)0.0f;
#pragma unroll
  for (int kt = 0; kt < 3; ++kt) {
    v16bf a = load_a_frag(sX, K1PAD, arow, kt * 32 + hi * 16);
#pragma unroll
    for (int nt = 0; nt < 4; ++nt) {
      v16bf bm = load_b_frag(sW, K1PAD, nt * 16 + lo, kt * 32 + hi * 16);
      acc[nt] = __builtin_amdgcn_wmma_f32_16x16x32_bf16(false, a, false, bm,
                                                        (short)0, acc[nt], false, false);
    }
  }
#pragma unroll
  for (int nt = 0; nt < 4; ++nt) {
    int ch = nt * 16 + lo;
    float s = 0.0f, s2 = 0.0f;
#pragma unroll
    for (int r = 0; r < 8; ++r) {
      float v = acc[nt][r];
      size_t row = m0 + wave * 16 + hi * 8 + r;
      h1[row * C1 + ch] = v;
      s += v; s2 += v * v;
    }
    atomicAdd(&sSum[ch], s);
    atomicAdd(&sSq[ch], s2);
  }
  __syncthreads();
  if (tid < C1) {
    atomicAdd(&stats[tid], sSum[tid]);
    atomicAdd(&stats[C1 + tid], sSq[tid]);
  }
}

// ------------------------------------- GEMM with fused BN+ReLU staging (K=64, NOUT=64/128)
template <int NOUT>
__global__ __launch_bounds__(256) void gemm_bn_kernel(const float* __restrict__ xin,
                                                      const float* __restrict__ ss,
                                                      const float* __restrict__ w,
                                                      float* __restrict__ hout,
                                                      float* __restrict__ stats) {
  __shared__ __attribute__((aligned(32))) __bf16 sX[128 * 64];
  __shared__ __attribute__((aligned(32))) __bf16 sW[NOUT * 64];
  __shared__ float sSum[NOUT], sSq[NOUT];
  const int tid = threadIdx.x;
  const size_t m0 = (size_t)blockIdx.x * 128;

  for (int i = tid; i < 128 * 64; i += 256) {           // BN + ReLU fused into staging
    int c = i & 63;
    float v = xin[m0 * 64 + i];
    sX[i] = (__bf16)fmaxf(v * ss[c] + ss[64 + c], 0.0f);
  }
  for (int i = tid; i < NOUT * 64; i += 256) sW[i] = (__bf16)w[i];
  if (tid < NOUT) { sSum[tid] = 0.0f; sSq[tid] = 0.0f; }
  if (tid == 0 && blockIdx.x + 1 < gridDim.x)
    __builtin_prefetch(xin + (m0 + 128) * 64, 0, 0);
  __syncthreads();

  const int lane = tid & 31, wave = tid >> 5;
  const int lo = lane & 15, hi = lane >> 4;
  const int arow = wave * 16 + lo;
  constexpr int NT = NOUT / 16;
  v8f acc[NT];
#pragma unroll
  for (int nt = 0; nt < NT; ++nt) acc[nt] = (v8f)0.0f;
#pragma unroll
  for (int kt = 0; kt < 2; ++kt) {
    v16bf a = load_a_frag(sX, 64, arow, kt * 32 + hi * 16);
#pragma unroll
    for (int nt = 0; nt < NT; ++nt) {
      v16bf bm = load_b_frag(sW, 64, nt * 16 + lo, kt * 32 + hi * 16);
      acc[nt] = __builtin_amdgcn_wmma_f32_16x16x32_bf16(false, a, false, bm,
                                                        (short)0, acc[nt], false, false);
    }
  }
#pragma unroll
  for (int nt = 0; nt < NT; ++nt) {
    int ch = nt * 16 + lo;
    float s = 0.0f, s2 = 0.0f;
#pragma unroll
    for (int r = 0; r < 8; ++r) {
      float v = acc[nt][r];
      size_t row = m0 + wave * 16 + hi * 8 + r;
      hout[row * NOUT + ch] = v;
      s += v; s2 += v * v;
    }
    atomicAdd(&sSum[ch], s);
    atomicAdd(&sSq[ch], s2);
  }
  __syncthreads();
  if (tid < NOUT) {
    atomicAdd(&stats[tid], sSum[tid]);
    atomicAdd(&stats[NOUT + tid], sSq[tid]);
  }
}

// ---------------------------------------------------------------- BN finalize
__global__ void finalize_kernel(const float* __restrict__ stats,
                                const float* __restrict__ gamma,
                                const float* __restrict__ beta,
                                float* __restrict__ ss, int C) {
  int c = threadIdx.x;
  if (c < C) {
    const float cnt = (float)MROWS;
    float mean = stats[c] / cnt;
    float var  = stats[C + c] / cnt - mean * mean;
    float sc   = gamma[c] * rsqrtf(var + 1e-5f);
    ss[c]     = sc;
    ss[C + c] = beta[c] - mean * sc;
  }
}

__global__ void zero_stats_kernel(float* __restrict__ stats) {
  stats[threadIdx.x] = 0.0f;   // 512 floats: 128+128+256
}

// ---------------------------------------------------------- max over S + BN3 + transpose
__global__ __launch_bounds__(256) void maxpool_kernel(const float* __restrict__ h3,
                                                      const float* __restrict__ ss3,
                                                      float* __restrict__ out_pts) {
  int idx = blockIdx.x * 256 + threadIdx.x;   // over NB*NP*C3
  int c = idx & (C3 - 1);
  int m = idx >> 7;
  int b = m >> 10, p = m & (NP - 1);
  const float* base = h3 + (size_t)m * NS * C3 + c;
  float mx = -3.4e38f;
#pragma unroll
  for (int s = 0; s < NS; ++s) mx = fmaxf(mx, base[(size_t)s * C3]);
  out_pts[((size_t)b * C3 + c) * NP + p] = mx * ss3[c] + ss3[C3 + c];  // BN after max (gamma>0)
}

// ---------------------------------------------------------------- launch
extern "C" void kernel_launch(void* const* d_in, const int* in_sizes, int n_in,
                              void* d_out, int out_size, void* d_ws, size_t ws_size,
                              hipStream_t stream) {
  (void)in_sizes; (void)n_in; (void)out_size; (void)ws_size;
  const float* xyz    = (const float*)d_in[0];
  const float* points = (const float*)d_in[1];
  const float* w1  = (const float*)d_in[2];
  const float* g1  = (const float*)d_in[4];
  const float* bt1 = (const float*)d_in[5];
  const float* w2  = (const float*)d_in[6];
  const float* g2  = (const float*)d_in[8];
  const float* bt2 = (const float*)d_in[9];
  const float* w3  = (const float*)d_in[10];
  const float* g3  = (const float*)d_in[12];
  const float* bt3 = (const float*)d_in[13];

  char* ws = (char*)d_ws;
  float*  new_xyz_t = (float*)(ws + 0);                 // 98304 B
  float*  stats1    = (float*)(ws + 98304);             // 128 f
  float*  stats2    = stats1 + 128;                     // 128 f
  float*  stats3    = stats2 + 128;                     // 256 f
  float*  ss1       = (float*)(ws + 100352);            // 128 f
  float*  ss2       = ss1 + 128;                        // 128 f
  float*  ss3       = ss2 + 128;                        // 256 f
  __bf16* feat = (__bf16*)(ws + 131072);                // M*96 bf16   (~48 MB)
  float*  h1   = (float*)(ws + 50462720ULL);            // M*64 f32    (~64 MB)
  float*  h2   = (float*)(ws + 117571584ULL);           // M*64 f32    (~64 MB)
  float*  h3   = (float*)(ws + 184680448ULL);           // M*128 f32   (~128 MB)

  float* out_xyz = (float*)d_out;                       // [B,3,P]
  float* out_pts = out_xyz + (size_t)NB * 3 * NP;       // [B,128,P]

  zero_stats_kernel<<<1, 512, 0, stream>>>(stats1);
  fps_kernel<<<NB, 1024, 0, stream>>>(xyz, out_xyz, new_xyz_t);
  knn_gather_kernel<<<NB * (NP / 256), 256, 0, stream>>>(xyz, points, new_xyz_t, feat);

  const int gblk = MROWS / 128;  // 2048
  gemm1_kernel<<<gblk, 256, 0, stream>>>(feat, w1, h1, stats1);
  finalize_kernel<<<1, 64, 0, stream>>>(stats1, g1, bt1, ss1, C1);
  gemm_bn_kernel<C2><<<gblk, 256, 0, stream>>>(h1, ss1, w2, h2, stats2);
  finalize_kernel<<<1, 64, 0, stream>>>(stats2, g2, bt2, ss2, C2);
  gemm_bn_kernel<C3><<<gblk, 256, 0, stream>>>(h2, ss2, w3, h3, stats3);
  finalize_kernel<<<1, 128, 0, stream>>>(stats3, g3, bt3, ss3, C3);
  maxpool_kernel<<<(NB * NP * C3) / 256, 256, 0, stream>>>(h3, ss3, out_pts);
}
